// RangeSpecializedMoERouter_67104569033199
// MI455X (gfx1250) — compile-verified, matrix operand order checked by reading, not run on hardware
//
#include <hip/hip_runtime.h>
#include <math.h>

typedef __attribute__((ext_vector_type(16))) _Float16 v16h;
typedef __attribute__((ext_vector_type(8)))  _Float16 v8h;
typedef __attribute__((ext_vector_type(8)))  float    v8f;
typedef __attribute__((ext_vector_type(4)))  float    v4f;

constexpr int Bn = 262144;
constexpr int Dd = 512;   // K dimension of GEMM1
constexpr int Hh = 64;    // N dimension of GEMM1
constexpr int Ee = 3;     // experts
constexpr int W1T_STRIDE = 520;           // halves; 1040B row stride -> 4-bank shift per row
constexpr int H_STRIDE   = 68;            // floats; 272B row stride -> conflict-free row reads
constexpr int ROWS_PER_BLOCK = 128;       // 8 waves x 16 rows
constexpr int NUM_TILES = Bn / ROWS_PER_BLOCK; // 2048

__global__ __launch_bounds__(256)
void moe_router_wmma(const float* __restrict__ x,
                     const int*   __restrict__ cit,
                     const float* __restrict__ W1,
                     const float* __restrict__ b1g,
                     const float* __restrict__ W2,
                     const float* __restrict__ b2g,
                     float* __restrict__ out,
                     float warm)
{
    __shared__ __align__(16) _Float16 sW1T[Hh * W1T_STRIDE];   // 65 KB f16 W1^T
    __shared__ __align__(16) float    sH[8][16 * H_STRIDE];    // 34 KB wave-private h tiles
    __shared__ float sW2T[Ee][Hh];
    __shared__ float sB1[Hh];
    __shared__ float sB2[Ee];

    const int tid  = threadIdx.x;
    const int wave = tid >> 5;     // wave32
    const int lane = tid & 31;

    // ---- One-time staging: W1^T as f16, W2^T, biases -------------------
    for (int i = tid; i < Dd * Hh; i += 256) {
        int k = i >> 6;        // 0..511
        int n = i & 63;        // 0..63
        sW1T[n * W1T_STRIDE + k] = (_Float16)W1[i];
    }
    for (int i = tid; i < Hh * Ee; i += 256) {
        sW2T[i % Ee][i / Ee] = W2[i];
    }
    if (tid < Hh) sB1[tid] = b1g[tid];
    if (tid < Ee) sB2[tid] = b2g[tid];
    __syncthreads();

    const float invT = 1.0f / (1.0f + 9.0f * warm);
    const float aLrn = (1.0f - warm) * invT;   // scale on learned logits
    const float aRt  = 10.0f * warm;           // scale on routing one-hot

    const int mrow  = lane & 15;               // M row within tile / N col of B frag
    const int kgrp  = lane >> 4;               // 0 or 1
    const int klo   = kgrp * 8;                // K half-group offset within a 32-K step
    const int mhalf = kgrp * 8;                // C-row offset (M = r + mhalf)

    for (int tile = blockIdx.x; tile < NUM_TILES; tile += gridDim.x) {
        const int rowBase = tile * ROWS_PER_BLOCK + wave * 16;
        const int row     = rowBase + mrow;
        const float* xp   = x + (size_t)row * Dd + klo;

        v8f acc[4] = {};   // 4 N-tiles of 16 -> H=64

        #pragma unroll 4
        for (int s = 0; s < 16; ++s) {
            // A fragment: 16-bit A 16x32 layout -> per lane K = klo..klo+7, klo+16..klo+23
            const float* px = xp + s * 32;
            v4f f0 = *(const v4f*)(px + 0);
            v4f f1 = *(const v4f*)(px + 4);
            v4f f2 = *(const v4f*)(px + 16);
            v4f f3 = *(const v4f*)(px + 20);
            v16h a;
            #pragma unroll
            for (int j = 0; j < 4; ++j) {
                a[j]      = (_Float16)f0[j];
                a[4 + j]  = (_Float16)f1[j];
                a[8 + j]  = (_Float16)f2[j];
                a[12 + j] = (_Float16)f3[j];
            }
            const int kbase = s * 32 + klo;
            #pragma unroll
            for (int t = 0; t < 4; ++t) {
                // B fragment (mirror of A's K pattern, lane column n = t*16 + mrow)
                const _Float16* wp = &sW1T[(t * 16 + mrow) * W1T_STRIDE + kbase];
                v8h lo = *(const v8h*)(wp);
                v8h hi = *(const v8h*)(wp + 16);
                v16h bfrag = __builtin_shufflevector(lo, hi,
                    0,1,2,3,4,5,6,7,8,9,10,11,12,13,14,15);
                acc[t] = __builtin_amdgcn_wmma_f32_16x16x32_f16(
                    false, a, false, bfrag, (short)0, acc[t], false, false);
            }
        }

        // Epilogue stage 1: + b1, relu, scatter C (lane=N, VGPR r = M) into LDS h tile
        float* hb = &sH[wave][0];
        #pragma unroll
        for (int t = 0; t < 4; ++t) {
            const int n   = t * 16 + mrow;
            const float bn = sB1[n];
            #pragma unroll
            for (int r = 0; r < 8; ++r) {
                float v = acc[t][r] + bn;
                hb[(r + mhalf) * H_STRIDE + n] = fmaxf(v, 0.0f);
            }
        }
        __syncthreads();

        // Epilogue stage 2: tiny GEMM (64x3), routing, softmax, store 3 floats/row
        if (lane < 16) {
            const int orow = rowBase + lane;
            const float* hr = &sH[wave][lane * H_STRIDE];
            float l0 = sB2[0], l1 = sB2[1], l2 = sB2[2];
            #pragma unroll 8
            for (int k = 0; k < Hh; ++k) {
                float hv = hr[k];
                l0 = fmaf(hv, sW2T[0][k], l0);
                l1 = fmaf(hv, sW2T[1][k], l1);
                l2 = fmaf(hv, sW2T[2][k], l2);
            }
            const int c = cit[orow];
            const int expert = (c <= 5) ? 0 : (c <= 50) ? 1 : 2;
            float c0 = aLrn * l0 + ((expert == 0) ? aRt : 0.0f);
            float c1 = aLrn * l1 + ((expert == 1) ? aRt : 0.0f);
            float c2 = aLrn * l2 + ((expert == 2) ? aRt : 0.0f);
            float m  = fmaxf(c0, fmaxf(c1, c2));
            float e0 = __expf(c0 - m);
            float e1 = __expf(c1 - m);
            float e2 = __expf(c2 - m);
            float inv = 1.0f / (e0 + e1 + e2);
            float* op = out + (size_t)orow * 3;
            op[0] = e0 * inv;
            op[1] = e1 * inv;
            op[2] = e2 * inv;
        }
        __syncthreads();
    }
}

extern "C" void kernel_launch(void* const* d_in, const int* in_sizes, int n_in,
                              void* d_out, int out_size, void* d_ws, size_t ws_size,
                              hipStream_t stream) {
    const float* x  = (const float*)d_in[0];
    const int*   cc = (const int*)  d_in[1];
    const float* W1 = (const float*)d_in[2];
    const float* b1 = (const float*)d_in[3];
    const float* W2 = (const float*)d_in[4];
    const float* b2 = (const float*)d_in[5];
    float* out = (float*)d_out;

    // warmup_coef = max(0, 1 - CURRENT_EPOCH/WARMUP_EPOCHS)^DECAY_RATE = 0.6^0.8
    float base = 1.0f - 2.0f / 5.0f;
    base = base > 0.0f ? base : 0.0f;
    float warm = powf(base, 0.8f);

    moe_router_wmma<<<dim3(512), dim3(256), 0, stream>>>(x, cc, W1, b1, W2, b2, out, warm);
}
